// HyperMultiheadAttention_31430570672467
// MI455X (gfx1250) — matrix operand order repaired and checked by reading, not compiled, not run on hardware
//
#include <hip/hip_runtime.h>

#define BB 4
#define NN 2048
#define DD 1024
#define HH 16
#define HD 64

typedef __attribute__((ext_vector_type(16))) __bf16 v16bf;
typedef __attribute__((ext_vector_type(8)))  float  v8f;

union FragU { uint4 q[2]; v16bf v; };

__device__ __forceinline__ unsigned short f2bf(float f) {
    unsigned int u = __float_as_uint(f);
    u += 0x7fffu + ((u >> 16) & 1u);          // round-to-nearest-even
    return (unsigned short)(u >> 16);
}

// ---- CDNA5 async copy: per-lane 16B global -> LDS, tracked by ASYNCcnt ----
__device__ __forceinline__ void async_load_b128(unsigned lds_byte_off, const void* gptr) {
    asm volatile("global_load_async_to_lds_b128 %0, %1, off"
                 :: "v"(lds_byte_off), "v"(gptr) : "memory");
}
__device__ __forceinline__ void wait_asynccnt_0() {
    asm volatile("s_wait_asynccnt 0x0" ::: "memory");
}
__device__ __forceinline__ void wait_asynccnt_1() {
    asm volatile("s_wait_asynccnt 0x1" ::: "memory");
}
__device__ __forceinline__ void wait_dscnt_0() {   // same-wave LDS ordering
    asm volatile("s_wait_dscnt 0x0" ::: "memory");
}

// A-fragment (16x32 bf16, MxK): lanes 0-15 row M=lane, K {0..7}+{16..23};
// lanes 16-31 same rows, K base +8 (ISA 7.12.2).
__device__ __forceinline__ v16bf load_a_frag(const unsigned short* base, int stride, int lane) {
    FragU u;
    const unsigned short* p = base + (lane & 15) * stride + ((lane >> 4) << 3);
    u.q[0] = *(const uint4*)(p);
    u.q[1] = *(const uint4*)(p + 16);
    return u.v;
}
// B-fragment (32x16 bf16, KxN) from an N-major [n][k] buffer:
// lanes 0-15 col N=lane, K {0..7}+{16..23}; lanes 16-31 K base +8.
__device__ __forceinline__ v16bf load_b_frag(const unsigned short* base, int stride, int lane) {
    FragU u;
    const unsigned short* p = base + (lane & 15) * stride + ((lane >> 4) << 3);
    u.q[0] = *(const uint4*)(p);
    u.q[1] = *(const uint4*)(p + 16);
    return u.v;
}
__device__ __forceinline__ v8f wmma_bf16(v16bf a, v16bf b, v8f c) {
    return __builtin_amdgcn_wmma_f32_16x16x32_bf16(false, a, false, b, (short)0, c, false, false);
}

// ---------------------------------------------------------------------------
// Tiled GEMM, BM=BN=128, BK=64, 8 waves (wave tile 32x64 = 8 accums).
// fp32 operands: register-staged double buffering + on-the-fly bf16 convert.
// bf16 A operand: async-DMA double-buffered LDS (ASYNCcnt).
// MODE 0: bf16 split-head (B,H,N,HD) out, scaled.  MODE 1: fp32 (B,N,D) out.
// ---------------------------------------------------------------------------
template <typename TA, int MODE>
__global__ __launch_bounds__(256)
void gemm_kernel(const TA* __restrict__ Aall, const float* __restrict__ Wall,
                 void* __restrict__ outp, float scale)
{
    constexpr bool AF32 = (sizeof(TA) == 4);
    constexpr int  NBUF = AF32 ? 1 : 2;
    __shared__ __align__(16) unsigned short As[NBUF][128 * 64];  // [m][k]
    __shared__ __align__(16) unsigned short Bs[128 * 64];        // [n][k]

    const int b = blockIdx.z;
    const int m0 = blockIdx.y * 128;
    const int n0 = blockIdx.x * 128;
    const int tid = threadIdx.x, lane = tid & 31, wid = tid >> 5;
    const int wm = (wid & 3) * 32, wn = (wid >> 2) * 64;

    const TA*    A = Aall + (size_t)b * NN * DD;
    const float* W = Wall + (size_t)b * DD * DD;

    float4 wr[8], ar[8];

    auto load_w_regs = [&](int kt) {
#pragma unroll
        for (int j = 0; j < 8; ++j) {
            int k = 8 * j + (tid >> 5), n = (tid & 31) << 2;
            wr[j] = *(const float4*)(W + (size_t)(kt + k) * DD + n0 + n);
        }
    };
    auto store_w_regs = [&]() {
#pragma unroll
        for (int j = 0; j < 8; ++j) {
            int k = 8 * j + (tid >> 5), n = (tid & 31) << 2;
            Bs[(n + 0) * 64 + k] = f2bf(wr[j].x);
            Bs[(n + 1) * 64 + k] = f2bf(wr[j].y);
            Bs[(n + 2) * 64 + k] = f2bf(wr[j].z);
            Bs[(n + 3) * 64 + k] = f2bf(wr[j].w);
        }
    };
    auto load_a_regs = [&](int kt) {
#pragma unroll
        for (int j = 0; j < 8; ++j) {
            int m = 16 * j + (tid >> 4), k = (tid & 15) << 2;
            ar[j] = *(const float4*)(A + (size_t)(m0 + m) * DD + kt + k);
        }
    };
    auto store_a_regs = [&]() {
#pragma unroll
        for (int j = 0; j < 8; ++j) {
            int m = 16 * j + (tid >> 4), k = (tid & 15) << 2;
            union { unsigned short s[4]; uint2 u; } t;
            t.s[0] = f2bf(ar[j].x); t.s[1] = f2bf(ar[j].y);
            t.s[2] = f2bf(ar[j].z); t.s[3] = f2bf(ar[j].w);
            *(uint2*)(&As[0][m * 64 + k]) = t.u;
        }
    };
    auto issue_a_async = [&](int kt, int nb) {      // bf16 A: verbatim DMA
#pragma unroll
        for (int j = 0; j < 2; ++j) {
            int m = 32 * j + (tid >> 3), k = (tid & 7) << 3;
            async_load_b128((unsigned)(size_t)&As[nb][m * 64 + k],
                            (const void*)(A + (size_t)(m0 + m) * DD + kt + k));
        }
    };

    // prologue: stage tile 0
    load_w_regs(0);
    if constexpr (AF32) load_a_regs(0);
    else                issue_a_async(0, 0);

    v8f acc[2][4] = {};
    constexpr int NT = DD / 64;                     // 16 K-tiles

    for (int t = 0; t < NT; ++t) {
        store_w_regs();
        if constexpr (AF32) store_a_regs();
        else                wait_asynccnt_0();      // tile t DMA landed
        __syncthreads();

        if (t + 1 < NT) {                           // overlap next stage w/ WMMA
            load_w_regs((t + 1) * 64);
            if constexpr (AF32) load_a_regs((t + 1) * 64);
            else                issue_a_async((t + 1) * 64, (t + 1) & 1);
        }

        const unsigned short* Asb = As[AF32 ? 0 : (t & 1)];
#pragma unroll
        for (int ks = 0; ks < 2; ++ks) {
            v16bf af[2], bfr[4];
#pragma unroll
            for (int mi = 0; mi < 2; ++mi)
                af[mi] = load_a_frag(Asb + (wm + mi * 16) * 64 + ks * 32, 64, lane);
#pragma unroll
            for (int ni = 0; ni < 4; ++ni)
                bfr[ni] = load_b_frag(Bs + (wn + ni * 16) * 64 + ks * 32, 64, lane);
#pragma unroll
            for (int mi = 0; mi < 2; ++mi)
#pragma unroll
                for (int ni = 0; ni < 4; ++ni)
                    acc[mi][ni] = wmma_bf16(af[mi], bfr[ni], acc[mi][ni]);
        }
        __syncthreads();
    }

    // epilogue (C-frag: row = r + 8*(lane>=16), col = lane&15)
    const int rlo = (lane >> 4) << 3, cl = lane & 15;
#pragma unroll
    for (int mi = 0; mi < 2; ++mi)
#pragma unroll
        for (int ni = 0; ni < 4; ++ni)
#pragma unroll
            for (int r = 0; r < 8; ++r) {
                int tok = m0 + wm + mi * 16 + rlo + r;
                int e   = n0 + wn + ni * 16 + cl;
                float v = acc[mi][ni][r] * scale;
                if constexpr (MODE == 0) {
                    int h = e >> 6, hd = e & 63;
                    ((unsigned short*)outp)[(((size_t)b * HH + h) * NN + tok) * HD + hd] = f2bf(v);
                } else {
                    ((float*)outp)[((size_t)b * NN + tok) * DD + e] = v;
                }
            }
}

// ---------------------------------------------------------------------------
// Flash attention: 128 queries/block (8 waves x 16 rows), 32 keys/iter.
// K tiles: async-DMA double-buffered LDS; V tiles: register double-buffered
// (transposed on store). Q pre-scaled by 1/sqrt(HD).
// ---------------------------------------------------------------------------
__global__ __launch_bounds__(256)
void attn_kernel(const unsigned short* __restrict__ Qb,
                 const unsigned short* __restrict__ Kb,
                 const unsigned short* __restrict__ Vb,
                 unsigned short* __restrict__ Ob)
{
    __shared__ __align__(16) unsigned short Ks[2][32 * 64];    // [key][hd]
    __shared__ __align__(16) unsigned short Vt[2][64 * 32];    // [hd][key]
    __shared__ __align__(16) unsigned short Ps[8][16 * 32];    // per-wave P bounce

    const int b = blockIdx.z, h = blockIdx.y;
    const int q0 = blockIdx.x * 128;
    const int tid = threadIdx.x, lane = tid & 31, wid = tid >> 5;
    const size_t bh = ((size_t)b * HH + h) * (size_t)NN * HD;
    const unsigned short* Q = Qb + bh;
    const unsigned short* K = Kb + bh;
    const unsigned short* V = Vb + bh;

    v16bf qa[2];
#pragma unroll
    for (int ks = 0; ks < 2; ++ks)
        qa[ks] = load_a_frag(Q + (size_t)(q0 + wid * 16) * HD + ks * 32, HD, lane);

    float m_r[8], l_r[8];
    v8f acc[4] = {};
#pragma unroll
    for (int r = 0; r < 8; ++r) { m_r[r] = -1e30f; l_r[r] = 0.0f; }

    union V8 { uint4 u; unsigned short s[8]; };
    V8 vcur, vnext; vnext.u = uint4{0, 0, 0, 0};
    const int key = tid >> 3, p8 = (tid & 7) << 3;

    // prologue: stage key-tile 0
    async_load_b128((unsigned)(size_t)&Ks[0][key * 64 + p8],
                    (const void*)(K + (size_t)key * HD + p8));
    vcur.u = *(const uint4*)(V + (size_t)key * HD + p8);

    constexpr int NTK = NN / 32;                    // 64 key tiles
    for (int t = 0; t < NTK; ++t) {
        const int buf = t & 1;
        // scatter V tile t (transposed) into LDS
#pragma unroll
        for (int j = 0; j < 8; ++j) Vt[buf][(p8 + j) * 32 + key] = vcur.s[j];
        if (t + 1 < NTK) {                          // prefetch tile t+1
            const int j1 = (t + 1) * 32;
            async_load_b128((unsigned)(size_t)&Ks[buf ^ 1][key * 64 + p8],
                            (const void*)(K + (size_t)(j1 + key) * HD + p8));
            vnext.u = *(const uint4*)(V + (size_t)(j1 + key) * HD + p8);
            wait_asynccnt_1();                      // tile t done (in-order)
        } else {
            wait_asynccnt_0();
        }
        __syncthreads();

        // S = Q K^T for 2x16 keys (2 WMMA k-steps over HD)
        v8f Sa = {}, Sb = {};
#pragma unroll
        for (int ks = 0; ks < 2; ++ks) {
            v16bf kb0 = load_b_frag(Ks[buf] + 0 * 64  + ks * 32, 64, lane);
            v16bf kb1 = load_b_frag(Ks[buf] + 16 * 64 + ks * 32, 64, lane);
            Sa = wmma_bf16(qa[ks], kb0, Sa);
            Sb = wmma_bf16(qa[ks], kb1, Sb);
        }

        // online softmax (row stats reduced across 16-lane groups)
        float f[8];
#pragma unroll
        for (int r = 0; r < 8; ++r) {
            float mx = fmaxf(Sa[r], Sb[r]);
            mx = fmaxf(mx, __shfl_xor(mx, 1));
            mx = fmaxf(mx, __shfl_xor(mx, 2));
            mx = fmaxf(mx, __shfl_xor(mx, 4));
            mx = fmaxf(mx, __shfl_xor(mx, 8));
            float mn = fmaxf(m_r[r], mx);
            f[r] = __expf(m_r[r] - mn);
            m_r[r] = mn;
        }
#pragma unroll
        for (int ni = 0; ni < 4; ++ni)
#pragma unroll
            for (int r = 0; r < 8; ++r) acc[ni][r] *= f[r];
#pragma unroll
        for (int r = 0; r < 8; ++r) {
            float pa = __expf(Sa[r] - m_r[r]);
            float pb = __expf(Sb[r] - m_r[r]);
            Sa[r] = pa; Sb[r] = pb;
            float s = pa + pb;
            s += __shfl_xor(s, 1);
            s += __shfl_xor(s, 2);
            s += __shfl_xor(s, 4);
            s += __shfl_xor(s, 8);
            l_r[r] = l_r[r] * f[r] + s;
        }

        // C-frag -> A-frag relayout through wave-private LDS (no block barrier:
        // same-wave DS ops are in-order; wait_dscnt fences the compiler/HW)
        {
            const int rlo = (lane >> 4) << 3, cl = lane & 15;
            unsigned short* P = Ps[wid];
#pragma unroll
            for (int r = 0; r < 8; ++r) {
                P[(rlo + r) * 32 + cl]      = f2bf(Sa[r]);
                P[(rlo + r) * 32 + 16 + cl] = f2bf(Sb[r]);
            }
        }
        wait_dscnt_0();

        v16bf pf = load_a_frag(Ps[wid], 32, lane);
#pragma unroll
        for (int ni = 0; ni < 4; ++ni) {
            v16bf vf = load_b_frag(Vt[buf] + (ni * 16) * 32, 32, lane);
            acc[ni] = wmma_bf16(pf, vf, acc[ni]);
        }
        __syncthreads();                            // guard Ks/Vt buffer reuse
        vcur = vnext;
    }

    // normalize + store merged-head bf16 (B,N,D)
    const int rlo = (lane >> 4) << 3, cl = lane & 15;
#pragma unroll
    for (int r = 0; r < 8; ++r) {
        float inv = 1.0f / l_r[r];
        int tok = q0 + wid * 16 + rlo + r;
#pragma unroll
        for (int ni = 0; ni < 4; ++ni) {
            int hd = ni * 16 + cl;
            Ob[((size_t)b * NN + tok) * DD + h * HD + hd] = f2bf(acc[ni][r] * inv);
        }
    }
}

extern "C" void kernel_launch(void* const* d_in, const int* in_sizes, int n_in,
                              void* d_out, int out_size, void* d_ws, size_t ws_size,
                              hipStream_t stream) {
    const float* x  = (const float*)d_in[0];
    const float* wq = (const float*)d_in[1];
    const float* wk = (const float*)d_in[2];
    const float* wv = (const float*)d_in[3];
    const float* wo = (const float*)d_in[4];

    unsigned short* ws = (unsigned short*)d_ws;
    const size_t QSZ = (size_t)BB * HH * NN * HD;
    unsigned short* qb = ws;
    unsigned short* kb = ws + QSZ;
    unsigned short* vb = ws + 2 * QSZ;
    unsigned short* ab = ws + 3 * QSZ;

    dim3 blk(256);
    dim3 gg(DD / 128, NN / 128, BB);   // (8, 16, 4)

    gemm_kernel<float, 0><<<gg, blk, 0, stream>>>(x, wq, qb, 0.125f);  // 1/sqrt(64)
    gemm_kernel<float, 0><<<gg, blk, 0, stream>>>(x, wk, kb, 1.0f);
    gemm_kernel<float, 0><<<gg, blk, 0, stream>>>(x, wv, vb, 1.0f);

    attn_kernel<<<dim3(NN / 128, HH, BB), blk, 0, stream>>>(qb, kb, vb, ab);

    gemm_kernel<unsigned short, 1><<<gg, blk, 0, stream>>>(ab, wo, d_out, 1.0f);
}
